// PTBlock_2551210574395
// MI455X (gfx1250) — compile-verified, matrix operand order checked by reading, not run on hardware
//
#include <hip/hip_runtime.h>
#include <hip/hip_bf16.h>

// Point Transformer block for MI455X (gfx1250, wave32, WMMA).
// Dominant cost: per-neighbor 128x128 MLPs (d2/g1/g2) = 25.8 GFLOP -> WMMA f16.
// One wave == one point's [K=16, C=128] tile; 16x16x32 WMMA, 96 wmma/point.
// Weights are staged per-workgroup into LDS (B frags = ds_load_b128), A-stage
// rows padded to 136 halves to avoid LDS bank conflicts.

#define Bn 4
#define Np 4096
#define Cc 128
#define Kk 16
#define RS 136          // padded A-stage row stride (halves): 272B = 68 words

typedef _Float16 h16;
typedef __attribute__((ext_vector_type(16))) _Float16 v16h;
typedef __attribute__((ext_vector_type(8)))  float    v8f;

// ---------------- workspace layout (bytes) ----------------
#define OFF_FRAG 0
#define OFF_BIAS 262144
#define OFF_WD1  266752
#define OFF_IDX  270336
#define OFF_PHI  1318912
#define OFF_PSI  (OFF_PHI + Bn*Np*Cc*4)
#define OFF_ALP  (OFF_PSI + Bn*Np*Cc*4)
#define OFF_Y    (OFF_ALP + Bn*Np*Cc*4)

// ---------------- WMMA fragment helpers ----------------
// A-matrix 16x32 f16 (ISA 7.12.2): lane l holds row m=l&15; halves h map to
// k = 16*(h>>3) + 8*(l>>4) + (h&7) -> two contiguous 16B chunks per lane
// (2x ds_load_b128); padded row stride staggers banks.
__device__ __forceinline__ v16h ldsA(const h16* st, int ks, int lane) {
  int m = lane & 15;
  int hb = (lane >> 4) << 3;
  const h16* row = st + m * RS + ks * 32;
  v16h a;
#pragma unroll
  for (int h = 0; h < 16; ++h) {
    int k = ((h >> 3) << 4) + hb + (h & 7);
    a[h] = row[k];
  }
  return a;
}

// Copy one 32KB f16 weight matrix (fragment layout) global -> LDS.
__device__ __forceinline__ void copyW(h16* dst, const h16* frag, int mat, int tid) {
  const uint4* s = (const uint4*)(frag + mat * 16384);
  uint4* d = (uint4*)dst;
#pragma unroll
  for (int i = 0; i < 8; ++i) d[tid + 256 * i] = s[tid + 256 * i];
}

// acc[nt] += stage[16x128] * W^T with W resident in LDS (16x128x128, 32 WMMAs)
__device__ __forceinline__ void gemm16(const h16* stage, const h16* ldsW,
                                       int lane, v8f* acc) {
#pragma unroll
  for (int ks = 0; ks < 4; ++ks) {
    v16h a = ldsA(stage, ks, lane);
#pragma unroll
    for (int nt = 0; nt < 8; ++nt) {
      v16h bf = *(const v16h*)(ldsW + ((nt * 4 + ks) * 32 + lane) * 16);
      acc[nt] = __builtin_amdgcn_wmma_f32_16x16x32_f16(
          false, a, false, bf, (short)0, acc[nt], false, false);
    }
  }
}

__device__ __forceinline__ v8f vzero8() { v8f z = {0,0,0,0,0,0,0,0}; return z; }

// ---------------- prep: fold BN, build f16 B-fragments ----------------
struct SqW { const float* W; const float* b; const float* s; const float* t; };
struct Prep { SqW m[8]; };  // 0 top,1 phi,2 psi,3 alpha,4 d2,5 g1,6 g2,7 down

__global__ void prep_frag_kernel(Prep P, h16* __restrict__ frag, float* __restrict__ bias) {
  int m = blockIdx.y;
  int e = blockIdx.x * blockDim.x + threadIdx.x;  // 0..16383
  int h = e & 15, lane = (e >> 4) & 31, ks = (e >> 9) & 3, nt = e >> 11;
  int o = nt * 16 + (lane & 15);
  int kidx = ((h >> 3) << 4) + ((lane >> 4) << 3) + (h & 7);
  int i = ks * 32 + kidx;
  float sc = P.m[m].s ? P.m[m].s[o] : 1.0f;
  frag[m * 16384 + e] = (h16)(P.m[m].W[o * Cc + i] * sc);
  if (e < Cc) {
    float sv = P.m[m].s ? P.m[m].s[e] : 1.0f;
    float tv = P.m[m].t ? P.m[m].t[e] : 0.0f;
    bias[m * Cc + e] = P.m[m].b[e] * sv + tv;
  }
}

__global__ void prep_d1_kernel(const float* __restrict__ W, const float* __restrict__ b,
                               const float* __restrict__ s, const float* __restrict__ t,
                               float* __restrict__ wd1, float* __restrict__ bd1) {
  int i = threadIdx.x;  // 0..511
  if (i < 384) {
    wd1[i] = W[i] * s[i / 3];
  } else if (i < 512) {
    int o = i - 384;
    bd1[o] = b[o] * s[o] + t[o];
  }
}

// ---------------- KNN: 1 thread / query, register top-16 ----------------
__global__ void knn_kernel(const float* __restrict__ p, int* __restrict__ idx) {
  int q = blockIdx.x * 256 + threadIdx.x;      // all 256 queries share batch
  int b = q >> 12, n = q & (Np - 1);
  const float* pb = p + b * 3 * Np;
  float qx = pb[n], qy = pb[Np + n], qz = pb[2 * Np + n];
  __shared__ float sx[256], sy[256], sz[256];
  float bd[Kk];
  int   bi[Kk];
#pragma unroll
  for (int t = 0; t < Kk; ++t) { bd[t] = 3.4e38f; bi[t] = 0; }
  for (int tile = 0; tile < Np; tile += 256) {
    __syncthreads();
    int j = tile + threadIdx.x;
    sx[threadIdx.x] = pb[j];
    sy[threadIdx.x] = pb[Np + j];
    sz[threadIdx.x] = pb[2 * Np + j];
    __syncthreads();
    for (int jj = 0; jj < 256; ++jj) {
      float dx = qx - sx[jj], dy = qy - sy[jj], dz = qz - sz[jj];
      float d = dx * dx + dy * dy + dz * dz;
      if (d < bd[Kk - 1]) {
        bd[Kk - 1] = d; bi[Kk - 1] = tile + jj;
#pragma unroll
        for (int t = Kk - 1; t > 0; --t) {
          if (bd[t] < bd[t - 1]) {
            float td = bd[t]; bd[t] = bd[t - 1]; bd[t - 1] = td;
            int ti = bi[t]; bi[t] = bi[t - 1]; bi[t - 1] = ti;
          }
        }
      }
    }
  }
#pragma unroll
  for (int t = 0; t < Kk; ++t) idx[q * Kk + t] = bi[t];
}

// ---------------- feat: top(BN) -> phi/psi/alpha, wave per 16-pt strip ----
// LDS: 8 x 4352 per-wave A-stage | 32KB shared weight buffer
__global__ void feat_kernel(const float* __restrict__ x, const h16* __restrict__ frag,
                            const float* __restrict__ bias,
                            float* __restrict__ phi, float* __restrict__ psi,
                            float* __restrict__ alp) {
  extern __shared__ char smem[];
  int tid = threadIdx.x, wid = tid >> 5, lane = tid & 31;
  h16* stage = (h16*)(smem + wid * 4352);
  h16* ldsW  = (h16*)(smem + 8 * 4352);
  int strip = blockIdx.x * 8 + wid;
  int b = strip >> 8;                 // 256 strips per batch
  int n0 = (strip & 255) << 4;
  const float* xb = x + b * Cc * Np;
#pragma unroll 8
  for (int h = 0; h < 64; ++h) {      // stage x^T tile [16 pts][128 ch] as f16
    int e = lane * 64 + h;
    int m = e >> 7, c = e & 127;
    stage[m * RS + c] = (h16)xb[c * Np + n0 + m];
  }
  copyW(ldsW, frag, 0, tid);
  __syncthreads();
  int hb = lane >> 4, col = lane & 15;

  v8f acc[8];
#pragma unroll
  for (int nt = 0; nt < 8; ++nt) acc[nt] = vzero8();
  gemm16(stage, ldsW, lane, acc);               // top
  __syncthreads();
  const float* bt = bias;
#pragma unroll
  for (int nt = 0; nt < 8; ++nt) {
    float bb = bt[nt * 16 + col];
#pragma unroll
    for (int r = 0; r < 8; ++r) {
      int m = r + (hb << 3);
      stage[m * RS + nt * 16 + col] = (h16)(acc[nt][r] + bb);
    }
  }
  copyW(ldsW, frag, 1, tid);
  __syncthreads();

#pragma unroll
  for (int L = 0; L < 3; ++L) {                 // phi, psi, alpha
#pragma unroll
    for (int nt = 0; nt < 8; ++nt) acc[nt] = vzero8();
    gemm16(stage, ldsW, lane, acc);
    const float* bl = bias + (1 + L) * Cc;
    float* o = (L == 0 ? phi : (L == 1 ? psi : alp)) + (b * Np + n0) * Cc;
#pragma unroll
    for (int nt = 0; nt < 8; ++nt) {
      float bb = bl[nt * 16 + col];
#pragma unroll
      for (int r = 0; r < 8; ++r) {
        int m = r + (hb << 3);
        o[m * Cc + nt * 16 + col] = acc[nt][r] + bb;
      }
    }
    if (L < 2) {                                // load next weight matrix
      __syncthreads();
      copyW(ldsW, frag, 2 + L, tid);
      __syncthreads();
    }
  }
}

// ---------------- attn: fused delta/gamma MLPs + softmax, wave per point ---
// LDS: 8 x 4608 per-wave (A-stage 4352 | pn 192 | nb 64) | 32KB weight buffer
__global__ void attn_kernel(const float* __restrict__ p, const int* __restrict__ idx,
                            const float* __restrict__ phi, const float* __restrict__ psi,
                            const float* __restrict__ alp,
                            const h16* __restrict__ frag, const float* __restrict__ bias,
                            const float* __restrict__ wd1, float* __restrict__ y) {
  extern __shared__ char smem[];
  int tid = threadIdx.x, wid = tid >> 5, lane = tid & 31;
  char* base = smem + wid * 4608;
  h16*   stage = (h16*)base;              // [16][RS] f16 A-stage
  float* pn    = (float*)(base + 4352);   // [16][3]
  int*   nb    = (int*)(base + 4544);     // [16]
  h16*   ldsW  = (h16*)(smem + 8 * 4608);
  int pt = blockIdx.x * 8 + wid;
  int b = pt >> 12, n = pt & (Np - 1);
  const float* pb = p + b * 3 * Np;
  if (lane < 16) {
    int j = idx[pt * Kk + lane];
    nb[lane] = j;
    pn[lane * 3 + 0] = pb[j];
    pn[lane * 3 + 1] = pb[Np + j];
    pn[lane * 3 + 2] = pb[2 * Np + j];
  }
  float qx = pb[n], qy = pb[Np + n], qz = pb[2 * Np + n];
  copyW(ldsW, frag, 4, tid);
  __syncthreads();

  // h = relu(BN(W_d1 @ rel)) -> f16 A-stage
  const float* bd1 = bias + 8 * Cc;
#pragma unroll 8
  for (int h = 0; h < 64; ++h) {
    int e = lane * 64 + h;
    int k = e >> 7, c = e & 127;
    float rx = qx - pn[k * 3], ry = qy - pn[k * 3 + 1], rz = qz - pn[k * 3 + 2];
    float hv = wd1[c * 3] * rx + wd1[c * 3 + 1] * ry + wd1[c * 3 + 2] * rz + bd1[c];
    stage[k * RS + c] = (h16)fmaxf(hv, 0.0f);
  }
  __syncthreads();
  int hb = lane >> 4, col = lane & 15;

  // delta = BN(W_d2 @ h), held in C-layout registers
  v8f dacc[8];
#pragma unroll
  for (int nt = 0; nt < 8; ++nt) dacc[nt] = vzero8();
  gemm16(stage, ldsW, lane, dacc);
  const float* bd2 = bias + 4 * Cc;
#pragma unroll
  for (int nt = 0; nt < 8; ++nt) {
    float bb = bd2[nt * 16 + col];
#pragma unroll
    for (int r = 0; r < 8; ++r) dacc[nt][r] += bb;
  }
  __syncthreads();

  // g_in = phi - psi_n + delta -> restage f16 ; also swap in W_g1
  const float* phv = phi + pt * Cc;
#pragma unroll
  for (int nt = 0; nt < 8; ++nt) {
    int c = nt * 16 + col;
    float pv = phv[c];
#pragma unroll
    for (int r = 0; r < 8; ++r) {
      int k = r + (hb << 3);
      float ps = psi[(b * Np + nb[k]) * Cc + c];
      stage[k * RS + c] = (h16)(pv - ps + dacc[nt][r]);
    }
  }
  copyW(ldsW, frag, 5, tid);
  __syncthreads();

  v8f gacc[8];
#pragma unroll
  for (int nt = 0; nt < 8; ++nt) gacc[nt] = vzero8();
  gemm16(stage, ldsW, lane, gacc);              // g1
  const float* bg1 = bias + 5 * Cc;
  __syncthreads();
#pragma unroll
  for (int nt = 0; nt < 8; ++nt) {
    float bb = bg1[nt * 16 + col];
#pragma unroll
    for (int r = 0; r < 8; ++r) {
      int k = r + (hb << 3);
      stage[k * RS + nt * 16 + col] = (h16)fmaxf(gacc[nt][r] + bb, 0.0f);
    }
  }
  copyW(ldsW, frag, 6, tid);
  __syncthreads();
#pragma unroll
  for (int nt = 0; nt < 8; ++nt) gacc[nt] = vzero8();
  gemm16(stage, ldsW, lane, gacc);              // g2
  const float* bg2 = bias + 6 * Cc;
#pragma unroll
  for (int nt = 0; nt < 8; ++nt) {
    float bb = bg2[nt * 16 + col];
#pragma unroll
    for (int r = 0; r < 8; ++r) gacc[nt][r] += bb;
  }

  // softmax over K (rows split lane vs lane^16) + aggregate
  float* yv = y + pt * Cc;
#pragma unroll
  for (int nt = 0; nt < 8; ++nt) {
    int c = nt * 16 + col;
    float mx = gacc[nt][0];
#pragma unroll
    for (int r = 1; r < 8; ++r) mx = fmaxf(mx, gacc[nt][r]);
    mx = fmaxf(mx, __shfl_xor(mx, 16, 32));
    float ex[8], s = 0.0f;
#pragma unroll
    for (int r = 0; r < 8; ++r) { ex[r] = __expf(gacc[nt][r] - mx); s += ex[r]; }
    s += __shfl_xor(s, 16, 32);
    float inv = 1.0f / s;
    float part = 0.0f;
#pragma unroll
    for (int r = 0; r < 8; ++r) {
      int k = r + (hb << 3);
      float av = alp[(b * Np + nb[k]) * Cc + c];
      part += ex[r] * inv * (av + dacc[nt][r]);
    }
    part += __shfl_xor(part, 16, 32);
    if (hb == 0) yv[c] = part;
  }
}

// ---------------- down: GEMM + BN + residual, store [B,C,N] ----------------
__global__ void down_kernel(const float* __restrict__ yv, const float* __restrict__ x,
                            const h16* __restrict__ frag, const float* __restrict__ bias,
                            float* __restrict__ out) {
  extern __shared__ char smem[];
  int tid = threadIdx.x, wid = tid >> 5, lane = tid & 31;
  h16* stage = (h16*)(smem + wid * 4352);
  h16* ldsW  = (h16*)(smem + 8 * 4352);
  int strip = blockIdx.x * 8 + wid;
  int b = strip >> 8;
  int n0 = (strip & 255) << 4;
  const float* yb = yv + (b * Np + n0) * Cc;   // [16][128] contiguous
#pragma unroll 8
  for (int h = 0; h < 64; ++h) {
    int e = lane * 64 + h;
    stage[(e >> 7) * RS + (e & 127)] = (h16)yb[e];
  }
  copyW(ldsW, frag, 7, tid);
  __syncthreads();
  v8f acc[8];
#pragma unroll
  for (int nt = 0; nt < 8; ++nt) acc[nt] = vzero8();
  gemm16(stage, ldsW, lane, acc);
  const float* bdn = bias + 7 * Cc;
  int hb = lane >> 4, col = lane & 15;
#pragma unroll
  for (int nt = 0; nt < 8; ++nt) {
    int o = nt * 16 + col;
    float bb = bdn[o];
    const float* xr = x + (b * Cc + o) * Np + n0;
    float* orow = out + (b * Cc + o) * Np + n0;
#pragma unroll
    for (int r = 0; r < 8; ++r) {
      int m = r + (hb << 3);
      orow[m] = acc[nt][r] + bb + xr[m];       // +residual, transposed store
    }
  }
}

// ---------------- launch ----------------
extern "C" void kernel_launch(void* const* d_in, const int* in_sizes, int n_in,
                              void* d_out, int out_size, void* d_ws, size_t ws_size,
                              hipStream_t stream) {
  const float* p = (const float*)d_in[0];
  const float* x = (const float*)d_in[1];
  const float* W_top = (const float*)d_in[2];  const float* b_top = (const float*)d_in[3];
  const float* W_phi = (const float*)d_in[4];  const float* b_phi = (const float*)d_in[5];
  const float* W_psi = (const float*)d_in[6];  const float* b_psi = (const float*)d_in[7];
  const float* W_alp = (const float*)d_in[8];  const float* b_alp = (const float*)d_in[9];
  const float* W_d1  = (const float*)d_in[10]; const float* b_d1  = (const float*)d_in[11];
  const float* W_d2  = (const float*)d_in[12]; const float* b_d2  = (const float*)d_in[13];
  const float* W_g1  = (const float*)d_in[14]; const float* b_g1  = (const float*)d_in[15];
  const float* W_g2  = (const float*)d_in[16]; const float* b_g2  = (const float*)d_in[17];
  const float* W_dn  = (const float*)d_in[18]; const float* b_dn  = (const float*)d_in[19];
  const float* s_top = (const float*)d_in[20]; const float* t_top = (const float*)d_in[21];
  const float* s_d1  = (const float*)d_in[22]; const float* t_d1  = (const float*)d_in[23];
  const float* s_d2  = (const float*)d_in[24]; const float* t_d2  = (const float*)d_in[25];
  const float* s_g1  = (const float*)d_in[26]; const float* t_g1  = (const float*)d_in[27];
  const float* s_g2  = (const float*)d_in[28]; const float* t_g2  = (const float*)d_in[29];
  const float* s_dn  = (const float*)d_in[30]; const float* t_dn  = (const float*)d_in[31];

  char* ws = (char*)d_ws;
  h16*   frag = (h16*)(ws + OFF_FRAG);
  float* bias = (float*)(ws + OFF_BIAS);
  float* wd1  = (float*)(ws + OFF_WD1);
  int*   idxb = (int*)(ws + OFF_IDX);
  float* phi  = (float*)(ws + OFF_PHI);
  float* psi  = (float*)(ws + OFF_PSI);
  float* alp  = (float*)(ws + OFF_ALP);
  float* yv   = (float*)(ws + OFF_Y);

  Prep P;
  P.m[0] = {W_top, b_top, s_top, t_top};
  P.m[1] = {W_phi, b_phi, nullptr, nullptr};
  P.m[2] = {W_psi, b_psi, nullptr, nullptr};
  P.m[3] = {W_alp, b_alp, nullptr, nullptr};
  P.m[4] = {W_d2,  b_d2,  s_d2,  t_d2};
  P.m[5] = {W_g1,  b_g1,  s_g1,  t_g1};
  P.m[6] = {W_g2,  b_g2,  s_g2,  t_g2};
  P.m[7] = {W_dn,  b_dn,  s_dn,  t_dn};

  prep_frag_kernel<<<dim3(64, 8), 256, 0, stream>>>(P, frag, bias);
  prep_d1_kernel<<<1, 512, 0, stream>>>(W_d1, b_d1, s_d1, t_d1, wd1, bias + 8 * Cc);
  knn_kernel<<<(Bn * Np) / 256, 256, 0, stream>>>(p, idxb);
  feat_kernel<<<(Bn * Np / 16) / 8, 256, 8 * 4352 + 32768, stream>>>(
      x, frag, bias, phi, psi, alp);
  attn_kernel<<<(Bn * Np) / 8, 256, 8 * 4608 + 32768, stream>>>(
      p, idxb, phi, psi, alp, frag, bias, wd1, yv);
  down_kernel<<<(Bn * Np / 16) / 8, 256, 8 * 4352 + 32768, stream>>>(
      yv, x, frag, bias, (float*)d_out);
}
